// NextStepRNN_51290499448985
// MI455X (gfx1250) — matrix operand order repaired and checked by reading, not compiled, not run on hardware
//
#include <hip/hip_runtime.h>
#include <math.h>

// ---------------------------------------------------------------------------
// NextStepRNN on MI455X (gfx1250, wave32, WMMA)
//
// Batch rows are independent across the scan, so the whole T=1024 recurrence
// runs in 8 persistent workgroups (16 batch rows each = one WMMA M-tile),
// synchronized only by intra-WGP barriers. All GEMMs use
// v_wmma_f32_16x16x32_bf16 (bf16 A/B, f32 accumulate); the input projection
// is fused into the recurrent GEMM by concatenating [obs | act | h] into one
// K=672 LDS A-panel against a pre-transposed bf16 weight block
// [W_in^T ; W_act^T ; W_rec^T] (672 x 512) in d_ws. Weights stream from L2
// (~800KB bf16, L2-resident) every step; an opaque zero offset (laundered
// through asm inside the t-loop) keeps those loads in-loop on the GLOBAL
// path without spilling. GEMM1 runs K-outer so each A-tile feeds 4
// back-to-back WMMAs and the next K-tile's loads overlap the current WMMAs.
// Next-step obs/act/noise (the cold HBM-resident inputs) are prefetched via
// global_prefetch_b8 one step ahead so staging never exposes HBM latency on
// the serial critical path.
// ---------------------------------------------------------------------------

#define T_STEPS 1024
#define BATCH   128
#define OBSD    128
#define ACTD    32
#define HDIM    512
#define KIN     (OBSD + ACTD)        // 160
#define KTOT    (OBSD + ACTD + HDIM) // 672  (21 K-tiles of 32)
#define EPSV    0.01f

#define ASTRIDE 680   // bf16 row stride of A panel: 1360B, 16B aligned, no bank conflicts
#define PSTRIDE 516   // f32 row stride of pre panel: conflict-free tile stores

typedef __attribute__((ext_vector_type(16))) __bf16 bf16x16;
typedef __attribute__((ext_vector_type(8)))  __bf16 bf16x8;
typedef __attribute__((ext_vector_type(8)))  float  f32x8;

// ---------------------------------------------------------------------------
// Weight prep: build bf16 transposed weight block in workspace.
//   BigBT[672][512]: rows 0..127 = W_in^T, 128..159 = W_act^T, 160..671 = W_rec^T
//   WoutT[512][128]: W_out^T
// So every WMMA B-operand load is: lane L -> col (nbase+(L&15)), 16 K rows at stride ld.
// ---------------------------------------------------------------------------
__global__ void __launch_bounds__(256) rnn_prep_weights(
    const float* __restrict__ W_in, const float* __restrict__ W_act,
    const float* __restrict__ W_rec, const float* __restrict__ W_out,
    __bf16* __restrict__ BigBT, __bf16* __restrict__ WoutT)
{
  int idx = blockIdx.x * 256 + threadIdx.x;
  const int nBig = KTOT * HDIM;              // 344064
  if (idx < nBig) {
    int j = idx / HDIM, i = idx - j * HDIM;  // B[k=j][n=i]
    float v;
    if (j < OBSD)       v = W_in[i * OBSD + j];
    else if (j < KIN)   v = W_act[i * ACTD + (j - OBSD)];
    else                v = W_rec[i * HDIM + (j - KIN)];
    BigBT[idx] = (__bf16)v;
  } else {
    int k = idx - nBig;                      // 65536 elements
    if (k < HDIM * OBSD) {
      int j = k / OBSD, o = k - j * OBSD;
      WoutT[k] = (__bf16)W_out[o * HDIM + j];
    }
  }
}

// A operand (16-bit, 16x32): lane L -> row L&15; slots 0..7 = K kbase+koff+0..7,
// slots 8..15 = K kbase+16+koff+0..7, koff = 8*(L>=16). Two 16B LDS loads.
__device__ __forceinline__ bf16x16 load_a(const __bf16* panel, int kbase, int lane) {
  const __bf16* p = panel + (size_t)(lane & 15) * ASTRIDE + kbase + ((lane >> 4) << 3);
  union { bf16x16 v; bf16x8 g[2]; } u;
  u.g[0] = *(const bf16x8*)(p);
  u.g[1] = *(const bf16x8*)(p + 16);
  return u.v;
}

// B operand (16-bit, 32x16): lane L -> col nbase+(L&15); slots 0..7 = K kbase+koff+0..7,
// slots 8..15 = K kbase+16+koff+0..7, koff = 8*(L>=16) (same K order as load_a).
__device__ __forceinline__ bf16x16 load_b(const __bf16* Wt, int kbase, int nbase,
                                          int ld, int lane) {
  const __bf16* p = Wt + (size_t)(kbase + ((lane >> 4) << 3)) * ld + nbase + (lane & 15);
  bf16x16 v;
#pragma unroll
  for (int s = 0; s < 8; ++s) {
    v[s]     = p[(size_t)s * ld];
    v[s + 8] = p[(size_t)(s + 16) * ld];
  }
  return v;
}

__device__ __forceinline__ f32x8 wmma_bf16(bf16x16 a, bf16x16 b, f32x8 c) {
  return __builtin_amdgcn_wmma_f32_16x16x32_bf16(false, a, false, b,
                                                 (short)0, c, false, false);
}

__global__ void __launch_bounds__(256, 1) rnn_scan(
    const float* __restrict__ obs, const float* __restrict__ act,
    const float* __restrict__ noise, const float* __restrict__ b_out,
    const float* __restrict__ beta_p, const float* __restrict__ nr_bias,
    const __bf16* __restrict__ BigBT_in, const __bf16* __restrict__ WoutT_in,
    float* __restrict__ out)
{
  __shared__ __bf16 A1[16 * ASTRIDE];           // [obs|act|h] panel, 21.3KB
  __shared__ float  preS[16 * PSTRIDE];         // pre-activation, 33KB
  __shared__ float  part_s[16][16], part_q[16][16];
  __shared__ float  mu_s[16], rinv_s[16];
  __shared__ float  biasS[HDIM];
  __shared__ float  boutS[OBSD];

  const int tid  = threadIdx.x;
  const int lane = tid & 31;
  const int wave = tid >> 5;                    // 0..7
  const int b0   = blockIdx.x * 16;             // batch-row slice

  for (int i = tid; i < HDIM; i += 256) biasS[i] = nr_bias[i];
  for (int i = tid; i < OBSD; i += 256) boutS[i] = b_out[i];
  for (int i = tid; i < 16 * HDIM; i += 256) {  // h0 = 0
    int r = i >> 9, c = i & (HDIM - 1);
    A1[r * ASTRIDE + KIN + c] = (__bf16)0.0f;
  }
  const float beta = beta_p[0];
  // Warm the first step's inputs while waiting at the first barrier.
  {
    const char* pnz = (const char*)(noise + (size_t)b0 * HDIM);
    __builtin_prefetch(pnz + tid * 128, 0, 0);
    if (tid < 64) {
      const char* pob = (const char*)(obs + (size_t)b0 * OBSD);
      __builtin_prefetch(pob + tid * 128, 0, 0);
    } else if (tid < 80) {
      const char* pac = (const char*)(act + (size_t)b0 * ACTD);
      __builtin_prefetch(pac + (tid - 64) * 128, 0, 0);
    }
  }
  __syncthreads();

  for (int t = 0; t < T_STEPS; ++t) {
    // Opaque zero offset, re-created every iteration: the compiler cannot
    // prove the weight loads are t-invariant (no hoist -> no 670-VGPR spill),
    // but pointer provenance is preserved so loads stay GLOBAL, not FLAT.
    size_t w0 = 0;
    asm volatile("" : "+s"(w0));
    const __bf16* __restrict__ BigBT = BigBT_in + w0;
    const __bf16* __restrict__ WoutT = WoutT_in + w0;

    // ---- stage obs/act tile as bf16 into A panel cols [0,160) ----
    {
      const float* ob = obs + ((size_t)t * BATCH + b0) * OBSD;
      const float* ac = act + ((size_t)t * BATCH + b0) * ACTD;
      for (int i = tid; i < 16 * KIN; i += 256) {
        int r = i / KIN, c = i - r * KIN;
        float v = (c < OBSD) ? ob[r * OBSD + c] : ac[r * ACTD + (c - OBSD)];
        A1[r * ASTRIDE + c] = (__bf16)v;
      }
    }

    // ---- prefetch next step's cold inputs (obs/act/noise) into WGP$/L2 ----
    // 128B lines; per WG: noise 32KB = 256 lines (one per thread),
    // obs 8KB = 64 lines, act 2KB = 16 lines. Overlaps GEMM1; no counters.
    if (t + 1 < T_STEPS) {
      const size_t tb1 = (size_t)(t + 1) * BATCH + b0;
      const char* pnz = (const char*)(noise + tb1 * HDIM);
      __builtin_prefetch(pnz + tid * 128, 0, 0);
      if (tid < 64) {
        const char* pob = (const char*)(obs + tb1 * OBSD);
        __builtin_prefetch(pob + tid * 128, 0, 0);
      } else if (tid < 80) {
        const char* pac = (const char*)(act + tb1 * ACTD);
        __builtin_prefetch(pac + (tid - 64) * 128, 0, 0);
      }
    }
    __syncthreads();

    // ---- GEMM1: pre(16x512) = [obs|act|h] @ [W_in^T;W_act^T;W_rec^T] + beta ----
    // K-outer: one A-tile feeds 4 N-tile WMMAs back-to-back; next K-tile's
    // 5 loads (1 LDS A + 4 global B) overlap the current 4 WMMAs.
    {
      const int n0 = wave * 16;                 // N-tiles: n0, n0+128, +256, +384
      f32x8 acc0, acc1, acc2, acc3;
      #pragma unroll
      for (int r = 0; r < 8; ++r) { acc0[r] = beta; acc1[r] = beta;
                                    acc2[r] = beta; acc3[r] = beta; }
      #pragma unroll 3
      for (int kb = 0; kb < KTOT; kb += 32) {
        bf16x16 aop = load_a(A1, kb, lane);
        bf16x16 bo0 = load_b(BigBT, kb, n0,       HDIM, lane);
        bf16x16 bo1 = load_b(BigBT, kb, n0 + 128, HDIM, lane);
        bf16x16 bo2 = load_b(BigBT, kb, n0 + 256, HDIM, lane);
        bf16x16 bo3 = load_b(BigBT, kb, n0 + 384, HDIM, lane);
        acc0 = wmma_bf16(aop, bo0, acc0);
        acc1 = wmma_bf16(aop, bo1, acc1);
        acc2 = wmma_bf16(aop, bo2, acc2);
        acc3 = wmma_bf16(aop, bo3, acc3);
      }
      // C/D layout: vgpr r -> row r + 8*(lane>=16), col ntile + (lane&15)
      const int col = lane & 15;
      const int mb  = (lane >> 4) << 3;
      #pragma unroll
      for (int r = 0; r < 8; ++r) {
        float* pr = &preS[(mb + r) * PSTRIDE + n0 + col];
        pr[0]   = acc0[r];
        pr[128] = acc1[r];
        pr[256] = acc2[r];
        pr[384] = acc3[r];
      }
    }
    __syncthreads();

    // ---- layernorm stats (mean + unbiased std), two-phase reduction ----
    {
      int r = tid >> 4, seg = tid & 15;
      const float* pr = &preS[r * PSTRIDE + seg * 32];
      float s = 0.f, q = 0.f;
      #pragma unroll
      for (int c = 0; c < 32; ++c) { float v = pr[c]; s += v; q += v * v; }
      part_s[r][seg] = s; part_q[r][seg] = q;
    }
    __syncthreads();
    if (tid < 16) {
      float s = 0.f, q = 0.f;
      #pragma unroll
      for (int i = 0; i < 16; ++i) { s += part_s[tid][i]; q += part_q[tid][i]; }
      float mu  = s * (1.0f / (float)HDIM);
      float var = (q - (float)HDIM * mu * mu) * (1.0f / (float)(HDIM - 1));
      var = fmaxf(var, 0.0f);
      mu_s[tid]   = mu;
      rinv_s[tid] = 1.0f / (sqrtf(var) + EPSV);
    }
    __syncthreads();

    // ---- activation: h = relu((pre-mu)/(std+eps) + nr_bias + noise) -> A panel ----
    {
      const float* nz = noise + ((size_t)t * BATCH + b0) * HDIM;
      for (int i = tid; i < 16 * HDIM; i += 256) {
        int r = i >> 9, c = i & (HDIM - 1);
        float v = (preS[r * PSTRIDE + c] - mu_s[r]) * rinv_s[r] + biasS[c] + nz[i];
        A1[r * ASTRIDE + KIN + c] = (__bf16)fmaxf(v, 0.0f);
      }
    }
    __syncthreads();

    // ---- GEMM2: y = sigmoid(h @ W_out^T + b_out), one 16x16 N-tile per wave ----
    {
      const int nbase = wave * 16;
      const int col   = nbase + (lane & 15);
      const int mb    = (lane >> 4) << 3;
      f32x8 acc;
      const float bo = boutS[col];
      #pragma unroll
      for (int r = 0; r < 8; ++r) acc[r] = bo;
      #pragma unroll 4
      for (int kb = 0; kb < HDIM; kb += 32) {
        bf16x16 aop = load_a(A1, KIN + kb, lane);
        bf16x16 bop = load_b(WoutT, kb, nbase, OBSD, lane);
        acc = wmma_bf16(aop, bop, acc);
      }
      float* op = out + ((size_t)t * BATCH + b0) * OBSD;
      #pragma unroll
      for (int r = 0; r < 8; ++r) {
        float y = 1.0f / (1.0f + __expf(-acc[r]));
        op[(size_t)(mb + r) * OBSD + col] = y;
      }
    }
    __syncthreads();
  }
}

extern "C" void kernel_launch(void* const* d_in, const int* in_sizes, int n_in,
                              void* d_out, int out_size, void* d_ws, size_t ws_size,
                              hipStream_t stream) {
  const float* obs     = (const float*)d_in[0];
  const float* act     = (const float*)d_in[1];
  const float* noise   = (const float*)d_in[2];
  const float* W_in    = (const float*)d_in[3];
  const float* W_act   = (const float*)d_in[4];
  const float* W_rec   = (const float*)d_in[5];
  const float* W_out   = (const float*)d_in[6];
  const float* b_out   = (const float*)d_in[7];
  const float* beta    = (const float*)d_in[8];
  const float* nr_bias = (const float*)d_in[9];
  float* out = (float*)d_out;

  __bf16* BigBT = (__bf16*)d_ws;                         // 672*512 bf16 = 672KB
  __bf16* WoutT = BigBT + (size_t)KTOT * HDIM;           // 512*128 bf16 = 128KB

  const int prep_elems = KTOT * HDIM + HDIM * OBSD;      // 409600
  rnn_prep_weights<<<(prep_elems + 255) / 256, 256, 0, stream>>>(
      W_in, W_act, W_rec, W_out, BigBT, WoutT);

  rnn_scan<<<BATCH / 16, 256, 0, stream>>>(
      obs, act, noise, b_out, beta, nr_bias, BigBT, WoutT, out);
}
